// SchNetInteraction_40295383171093
// MI455X (gfx1250) — matrix operand order repaired and checked by reading, not compiled, not run on hardware
//
#include <hip/hip_runtime.h>

typedef _Float16 v16h __attribute__((ext_vector_type(16)));
typedef _Float16 v8h  __attribute__((ext_vector_type(8)));
typedef float    v8f  __attribute__((ext_vector_type(8)));

#define HD  128   // node dim == hidden dim
#define ED  50    // edge feature dim
#define EDP 64    // padded K for edge layer 1

// ---------------------------------------------------------------- utilities

__global__ void zero_f32(float* __restrict__ p, long long n) {
    long long i = (long long)blockIdx.x * blockDim.x + threadIdx.x;
    long long stride = (long long)gridDim.x * blockDim.x;
    for (; i < n; i += stride) p[i] = 0.0f;
}

// Pack a [Kpad x N] (src is [Kact x N] row-major f32) weight matrix into
// WMMA B-fragment order: fragment(kt,nt) is 512 f16; within a fragment,
// lane L element j holds B[32*kt + (L<16?0:8) + (j<8?j:j+8)][16*nt + (L&15)].
// Each lane's 16 f16 are contiguous -> one 32B global load per fragment.
__global__ void pack_b(_Float16* __restrict__ dst, const float* __restrict__ src,
                       int Kact, int Kpad, int N) {
    int total = (Kpad / 32) * (N / 16) * 512;
    int stride = gridDim.x * blockDim.x;
    for (int gid = blockIdx.x * blockDim.x + threadIdx.x; gid < total; gid += stride) {
        int frag = gid >> 9;
        int r    = gid & 511;
        int lane = r >> 4;
        int j    = r & 15;
        int ntiles = N >> 4;
        int kt = frag / ntiles;
        int nt = frag - kt * ntiles;
        int k = kt * 32 + ((lane & 16) ? 8 : 0) + (j < 8 ? j : j + 8);
        int n = nt * 16 + (lane & 15);
        float v = (k < Kact) ? src[(long long)k * N + n] : 0.0f;
        dst[gid] = (_Float16)v;
    }
}

// SiLU with hardware rcp (single v_rcp_f32) instead of IEEE division.
__device__ __forceinline__ float silu(float v) {
    return v * __builtin_amdgcn_rcpf(1.0f + __expf(-v));
}

// Load a 16x32 f16 A-fragment from LDS. ISA layout: lane L holds row M=L&15;
// lanes 0-15 cover K = kb+0..7 and kb+16..23; lanes 16-31 cover +8 offsets.
__device__ __forceinline__ v16h lds_a_frag(const _Float16* __restrict__ base,
                                           int strideElems, int kbase,
                                           int lm, int lhk) {
    const _Float16* p = base + lm * strideElems + kbase + lhk;
    union { v16h v; v8h h[2]; } u;
    u.h[0] = *(const v8h*)p;
    u.h[1] = *(const v8h*)(p + 16);
    return u.v;
}

// ---------------------------------------------------------------- edge path
// 32 edges per workgroup iteration (two 16-row M-tiles), 4 waves each owning
// a 32-column slice of the 128 output columns.

__global__ void __launch_bounds__(128)
edge_kernel(const float* __restrict__ x,        // [N,128]
            const int*   __restrict__ eidx,     // [2,E]
            const float* __restrict__ ef,       // [E,50]
            const _Float16* __restrict__ We1p,  // 16 fragments (K=64 padded)
            const _Float16* __restrict__ We2p,  // 32 fragments (K=128)
            const float* __restrict__ be1,
            const float* __restrict__ be2,
            float* __restrict__ agg,            // [N,128] f32 accumulator
            int E, int numTiles)
{
    __shared__ __align__(16) _Float16 efl[32 * EDP];  // edge feats (f16, padded)
    __shared__ __align__(16) _Float16 hl[32 * HD];    // hidden acts (f16)
    __shared__ int sl[32], dl[32];

    const int tid  = threadIdx.x;
    const int lane = tid & 31;
    const int w    = tid >> 5;            // wave id 0..3 -> cols [32w, 32w+32)
    const int lm   = lane & 15;
    const int lhk  = (lane >> 4) << 3;    // 0 or 8: A K-half / C row-half offset

    // Preload B fragments into registers (reused for every edge tile).
    v16h b1[2][2], b2[4][2];
#pragma unroll
    for (int kt = 0; kt < 2; ++kt)
#pragma unroll
        for (int t = 0; t < 2; ++t)
            b1[kt][t] = *(const v16h*)(We1p + ((kt * 8 + (2 * w + t)) * 512 + lane * 16));
#pragma unroll
    for (int kt = 0; kt < 4; ++kt)
#pragma unroll
        for (int t = 0; t < 2; ++t)
            b2[kt][t] = *(const v16h*)(We2p + ((kt * 8 + (2 * w + t)) * 512 + lane * 16));

    float bias1[2], bias2[2];
#pragma unroll
    for (int t = 0; t < 2; ++t) {
        int col = 32 * w + 16 * t + lm;
        bias1[t] = be1[col];
        bias2[t] = be2[col];
    }

    for (int tile = blockIdx.x; tile < numTiles; tile += gridDim.x) {
        const long long e0 = (long long)tile * 32;

        // Stage 32 edges' features into LDS as f16 (K padded 50 -> 64).
        {
            int row = tid >> 2;            // 0..31
            int c0  = (tid & 3) * 16;      // 0,16,32,48
            const float* sp = ef + (e0 + row) * ED;
            bool rowOk = (e0 + row) < E;
#pragma unroll
            for (int j = 0; j < 16; ++j) {
                int c = c0 + j;
                float v = (rowOk && c < ED) ? sp[c] : 0.0f;
                efl[row * EDP + c] = (_Float16)v;
            }
            if (tid < 32) {
                bool ok = (e0 + tid) < E;
                sl[tid] = ok ? eidx[e0 + tid] : 0;
                dl[tid] = ok ? eidx[(long long)E + e0 + tid] : 0;
            }
        }
        __syncthreads();

        // Layer 1: h = silu(ef @ We1 + be1)  (this wave: cols [32w, 32w+32))
        v8f acc[2][2];                     // [mtile][ntile]
#pragma unroll
        for (int mt = 0; mt < 2; ++mt)
#pragma unroll
            for (int t = 0; t < 2; ++t)
#pragma unroll
                for (int r = 0; r < 8; ++r) acc[mt][t][r] = bias1[t];

#pragma unroll
        for (int mt = 0; mt < 2; ++mt)
#pragma unroll
            for (int kt = 0; kt < 2; ++kt) {
                v16h a = lds_a_frag(efl + mt * 16 * EDP, EDP, kt * 32, lm, lhk);
#pragma unroll
                for (int t = 0; t < 2; ++t)
                    acc[mt][t] = __builtin_amdgcn_wmma_f32_16x16x32_f16(
                        false, a, false, b1[kt][t], (short)0, acc[mt][t], false, false);
            }

#pragma unroll
        for (int mt = 0; mt < 2; ++mt)
#pragma unroll
            for (int t = 0; t < 2; ++t) {
                int col = 32 * w + 16 * t + lm;
#pragma unroll
                for (int r = 0; r < 8; ++r) {
                    int m = 16 * mt + r + lhk;
                    hl[m * HD + col] = (_Float16)silu(acc[mt][t][r]);
                }
            }
        __syncthreads();

        // Layer 2: ew = h @ We2 + be2
        v8f acc2[2][2];
#pragma unroll
        for (int mt = 0; mt < 2; ++mt)
#pragma unroll
            for (int t = 0; t < 2; ++t)
#pragma unroll
                for (int r = 0; r < 8; ++r) acc2[mt][t][r] = bias2[t];

#pragma unroll
        for (int mt = 0; mt < 2; ++mt)
#pragma unroll
            for (int kt = 0; kt < 4; ++kt) {
                v16h a = lds_a_frag(hl + mt * 16 * HD, HD, kt * 32, lm, lhk);
#pragma unroll
                for (int t = 0; t < 2; ++t)
                    acc2[mt][t] = __builtin_amdgcn_wmma_f32_16x16x32_f16(
                        false, a, false, b2[kt][t], (short)0, acc2[mt][t], false, false);
            }

        // messages = x[src] * ew ; scatter-add to agg[dst] (L2-resident atomics)
#pragma unroll
        for (int mt = 0; mt < 2; ++mt)
#pragma unroll
            for (int t = 0; t < 2; ++t) {
                int col = 32 * w + 16 * t + lm;
#pragma unroll
                for (int r = 0; r < 8; ++r) {
                    int m = 16 * mt + r + lhk;
                    if (e0 + m < E) {
                        float msg = acc2[mt][t][r] * x[(long long)sl[m] * HD + col];
                        atomicAdd(agg + (long long)dl[m] * HD + col, msg);
                    }
                }
            }
        __syncthreads();
    }
}

// ---------------------------------------------------------------- node path

__global__ void __launch_bounds__(128)
node_kernel(const float* __restrict__ x,        // [N,128]
            const float* __restrict__ agg,      // [N,128]
            const _Float16* __restrict__ Wn1p,  // 32 fragments
            const _Float16* __restrict__ Wn2p,  // 32 fragments
            const float* __restrict__ bn1,
            const float* __restrict__ bn2,
            float* __restrict__ out,            // [N,128]
            int Nn, int numTiles)
{
    __shared__ __align__(16) _Float16 al[32 * HD];
    __shared__ __align__(16) _Float16 hl[32 * HD];

    const int tid  = threadIdx.x;
    const int lane = tid & 31;
    const int w    = tid >> 5;
    const int lm   = lane & 15;
    const int lhk  = (lane >> 4) << 3;

    v16h b1[4][2], b2[4][2];
#pragma unroll
    for (int kt = 0; kt < 4; ++kt)
#pragma unroll
        for (int t = 0; t < 2; ++t) {
            b1[kt][t] = *(const v16h*)(Wn1p + ((kt * 8 + (2 * w + t)) * 512 + lane * 16));
            b2[kt][t] = *(const v16h*)(Wn2p + ((kt * 8 + (2 * w + t)) * 512 + lane * 16));
        }

    float bias1[2], bias2[2];
#pragma unroll
    for (int t = 0; t < 2; ++t) {
        int col = 32 * w + 16 * t + lm;
        bias1[t] = bn1[col];
        bias2[t] = bn2[col];
    }

    for (int tile = blockIdx.x; tile < numTiles; tile += gridDim.x) {
        const long long n0 = (long long)tile * 32;

        // Stage agg tile [32,128] -> LDS f16
        {
            int row = tid >> 2;            // 0..31
            int c0  = (tid & 3) * 32;      // 0,32,64,96
            const float* sp = agg + (n0 + row) * HD;
            bool rowOk = (n0 + row) < Nn;
#pragma unroll
            for (int j = 0; j < 32; ++j) {
                float v = rowOk ? sp[c0 + j] : 0.0f;
                al[row * HD + c0 + j] = (_Float16)v;
            }
        }
        __syncthreads();

        // Layer 1
        v8f acc[2][2];
#pragma unroll
        for (int mt = 0; mt < 2; ++mt)
#pragma unroll
            for (int t = 0; t < 2; ++t)
#pragma unroll
                for (int r = 0; r < 8; ++r) acc[mt][t][r] = bias1[t];
#pragma unroll
        for (int mt = 0; mt < 2; ++mt)
#pragma unroll
            for (int kt = 0; kt < 4; ++kt) {
                v16h a = lds_a_frag(al + mt * 16 * HD, HD, kt * 32, lm, lhk);
#pragma unroll
                for (int t = 0; t < 2; ++t)
                    acc[mt][t] = __builtin_amdgcn_wmma_f32_16x16x32_f16(
                        false, a, false, b1[kt][t], (short)0, acc[mt][t], false, false);
            }
#pragma unroll
        for (int mt = 0; mt < 2; ++mt)
#pragma unroll
            for (int t = 0; t < 2; ++t) {
                int col = 32 * w + 16 * t + lm;
#pragma unroll
                for (int r = 0; r < 8; ++r)
                    hl[(16 * mt + r + lhk) * HD + col] = (_Float16)silu(acc[mt][t][r]);
            }
        __syncthreads();

        // Layer 2 + residual
        v8f acc2[2][2];
#pragma unroll
        for (int mt = 0; mt < 2; ++mt)
#pragma unroll
            for (int t = 0; t < 2; ++t)
#pragma unroll
                for (int r = 0; r < 8; ++r) acc2[mt][t][r] = bias2[t];
#pragma unroll
        for (int mt = 0; mt < 2; ++mt)
#pragma unroll
            for (int kt = 0; kt < 4; ++kt) {
                v16h a = lds_a_frag(hl + mt * 16 * HD, HD, kt * 32, lm, lhk);
#pragma unroll
                for (int t = 0; t < 2; ++t)
                    acc2[mt][t] = __builtin_amdgcn_wmma_f32_16x16x32_f16(
                        false, a, false, b2[kt][t], (short)0, acc2[mt][t], false, false);
            }
#pragma unroll
        for (int mt = 0; mt < 2; ++mt)
#pragma unroll
            for (int t = 0; t < 2; ++t) {
                int col = 32 * w + 16 * t + lm;
#pragma unroll
                for (int r = 0; r < 8; ++r) {
                    long long node = n0 + 16 * mt + r + lhk;
                    if (node < Nn)
                        out[node * HD + col] = acc2[mt][t][r] + x[node * HD + col];
                }
            }
        __syncthreads();
    }
}

// ---------------------------------------------------------------- launcher

extern "C" void kernel_launch(void* const* d_in, const int* in_sizes, int n_in,
                              void* d_out, int out_size, void* d_ws, size_t ws_size,
                              hipStream_t stream) {
    const float* x   = (const float*)d_in[0];
    const int*  eidx = (const int*)  d_in[1];
    const float* ef  = (const float*)d_in[2];
    const float* We1 = (const float*)d_in[3];
    const float* be1 = (const float*)d_in[4];
    const float* We2 = (const float*)d_in[5];
    const float* be2 = (const float*)d_in[6];
    const float* Wn1 = (const float*)d_in[7];
    const float* bn1 = (const float*)d_in[8];
    const float* Wn2 = (const float*)d_in[9];
    const float* bn2 = (const float*)d_in[10];
    float* out = (float*)d_out;

    const int Nn = in_sizes[0] / HD;   // 50000
    const int E  = in_sizes[1] / 2;    // 800000

    // Workspace layout: [agg f32 N*128][We1p][We2p][Wn1p][Wn2p]
    char* ws = (char*)d_ws;
    float* agg = (float*)ws;
    size_t aggBytes = ((size_t)Nn * HD * sizeof(float) + 255) & ~(size_t)255;
    _Float16* We1p = (_Float16*)(ws + aggBytes);
    _Float16* We2p = We1p + 16 * 512;   // (64/32)*(128/16) = 16 fragments
    _Float16* Wn1p = We2p + 32 * 512;   // (128/32)*(128/16) = 32 fragments
    _Float16* Wn2p = Wn1p + 32 * 512;

    zero_f32<<<2048, 256, 0, stream>>>(agg, (long long)Nn * HD);
    pack_b<<<64, 256, 0, stream>>>(We1p, We1, ED, EDP, HD);
    pack_b<<<64, 256, 0, stream>>>(We2p, We2, HD, HD, HD);
    pack_b<<<64, 256, 0, stream>>>(Wn1p, Wn1, HD, HD, HD);
    pack_b<<<64, 256, 0, stream>>>(Wn2p, Wn2, HD, HD, HD);

    const int eTiles = (E + 31) / 32;
    const int nTiles = (Nn + 31) / 32;
    const int eGrid  = eTiles < 4096 ? eTiles : 4096;
    edge_kernel<<<eGrid, 128, 0, stream>>>(x, eidx, ef, We1p, We2p, be1, be2, agg, E, eTiles);
    node_kernel<<<nTiles, 128, 0, stream>>>(x, agg, Wn1p, Wn2p, bn1, bn2, out, Nn, nTiles);
}
